// LSTMX_11716670783838
// MI455X (gfx1250) — compile-verified, matrix operand order checked by reading, not run on hardware
//
#include <hip/hip_runtime.h>

#define BB 128   // batch
#define TT 256   // seq len
#define CC 64    // channels
#define HH 128   // hidden
#define GG 512   // 4*H
#define BSPLIT 2
#define BLOC (BB / BSPLIT)   // 64 batch rows per block
#define NTHREADS 512         // 16 wave32
#define NPAIRS ((BLOC / 16) * (HH / 16))   // 4 * 8 = 32
#define PAIRS_PER_WAVE (NPAIRS / (NTHREADS / 32))  // 2

typedef __attribute__((ext_vector_type(16))) _Float16 v16h;
typedef __attribute__((ext_vector_type(8)))  _Float16 v8h;
typedef __attribute__((ext_vector_type(8)))  float    v8f;

__device__ __forceinline__ float sigmoidf_(float x) {
    return 1.0f / (1.0f + __expf(-x));
}

// Load a 16x32 f16 WMMA operand fragment from row-major LDS storage.
// p points at element [rowbase + laneRow]*HH + kbase + 8*laneHalf.
// Per ISA layout: lane-half 0 holds K {kb..kb+7, kb+16..kb+23},
//                 lane-half 1 holds K {kb+8..kb+15, kb+24..kb+31}.
__device__ __forceinline__ v16h load_frag(const _Float16* p) {
    v8h lo = *(const v8h*)(p);
    v8h hi = *(const v8h*)(p + 16);
    v16h f;
#pragma unroll
    for (int j = 0; j < 8; ++j) { f[j] = lo[j]; f[j + 8] = hi[j]; }
    return f;
}

__global__ __launch_bounds__(NTHREADS, 1)
void lstm_chan_wmma(const float* __restrict__ x,     // [B,T,C]
                    const float* __restrict__ w_ih,  // [C,4H]
                    const float* __restrict__ w_hh,  // [C,4H,H]
                    const float* __restrict__ b_ih,  // [C,4H]
                    const float* __restrict__ b_hh,  // [C,4H]
                    const float* __restrict__ w_fc,  // [C,H]
                    const float* __restrict__ b_fc,  // [C]
                    float* __restrict__ out)         // [B,C]
{
    __shared__ __align__(16) _Float16 wlds[GG * HH];        // 128 KB f16 weights [n][k]
    __shared__ __align__(16) _Float16 hbuf[2][BLOC * HH];   // 2 x 16 KB f16 hidden (ping-pong)
    __shared__ __align__(16) float    clds[BLOC * HH];      // 32 KB f32 cell state
    __shared__ float blds[GG];                              // b_ih + b_hh
    __shared__ float wilds[GG];                             // w_ih
    __shared__ float xs[BLOC];

    const int c    = blockIdx.x / BSPLIT;
    const int bq   = blockIdx.x % BSPLIT;
    const int b0   = bq * BLOC;
    const int tid  = threadIdx.x;
    const int wv   = tid >> 5;
    const int lane = tid & 31;
    const int lh   = lane >> 4;   // lane half (0/1)
    const int ml   = lane & 15;   // lane row/col within half

    // ---- one-time init: stage channel weights into LDS (f32 -> f16) ----
    const float* whc = w_hh + (size_t)c * GG * HH;
    for (int i = tid * 4; i < GG * HH; i += NTHREADS * 4) {
        float4 v = *(const float4*)(whc + i);
        wlds[i + 0] = (_Float16)v.x;
        wlds[i + 1] = (_Float16)v.y;
        wlds[i + 2] = (_Float16)v.z;
        wlds[i + 3] = (_Float16)v.w;
    }
    if (tid < GG) {
        blds[tid]  = b_ih[(size_t)c * GG + tid] + b_hh[(size_t)c * GG + tid];
        wilds[tid] = w_ih[(size_t)c * GG + tid];
    }
    for (int i = tid; i < BLOC * HH; i += NTHREADS) {
        hbuf[0][i] = (_Float16)0.0f;
        clds[i]    = 0.0f;
    }
    __syncthreads();

    // ---- sequential recurrence over T ----
    for (int t = 0; t < TT; ++t) {
        const _Float16* hcur  = hbuf[t & 1];
        _Float16*       hnext = hbuf[(t + 1) & 1];

        if (tid < BLOC)
            xs[tid] = x[((size_t)(b0 + tid) * TT + t) * CC + c];
        __syncthreads();

#pragma unroll
        for (int i = 0; i < PAIRS_PER_WAVE; ++i) {
            const int p     = wv * PAIRS_PER_WAVE + i;
            const int mbase = (p >> 3) * 16;      // batch-tile base (0..48)
            const int hbase = (p & 7) * 16;       // hidden-col tile base (0..112)

            // A fragments: h tile [16 x 32] over K = 0..127
            v16h A[4];
#pragma unroll
            for (int kk = 0; kk < 4; ++kk)
                A[kk] = load_frag(hcur + (mbase + ml) * HH + kk * 32 + 8 * lh);

            // 4 gate tiles (i,f,g,o): B from wlds rows n = g*H + hbase + ml
            v8f acc[4];
#pragma unroll
            for (int g = 0; g < 4; ++g) {
                v8f a = {};
#pragma unroll
                for (int kk = 0; kk < 4; ++kk) {
                    v16h Bf = load_frag(wlds + (g * HH + hbase + ml) * HH + kk * 32 + 8 * lh);
                    a = __builtin_amdgcn_wmma_f32_16x16x32_f16(
                            /*neg_a=*/false, A[kk], /*neg_b=*/false, Bf,
                            /*c_mod=*/(short)0, a, /*reuse_a=*/false, /*reuse_b=*/false);
                }
                acc[g] = a;
            }

            // Elementwise gate math; accumulator elem r -> (b = mbase+8*lh+r, col = hbase+ml)
            const int hcol = hbase + ml;
            const float bi = blds[hcol],          wi = wilds[hcol];
            const float bf = blds[HH + hcol],     wf = wilds[HH + hcol];
            const float bg = blds[2 * HH + hcol], wg = wilds[2 * HH + hcol];
            const float bo = blds[3 * HH + hcol], wo = wilds[3 * HH + hcol];
#pragma unroll
            for (int r = 0; r < 8; ++r) {
                const int   b  = mbase + 8 * lh + r;
                const float xv = xs[b];
                const float gi = sigmoidf_(acc[0][r] + bi + xv * wi);
                const float gf = sigmoidf_(acc[1][r] + bf + xv * wf);
                const float gg = tanhf   (acc[2][r] + bg + xv * wg);
                const float go = sigmoidf_(acc[3][r] + bo + xv * wo);
                const float cold = clds[b * HH + hcol];
                const float cnew = gf * cold + gi * gg;
                clds[b * HH + hcol]  = cnew;
                hnext[b * HH + hcol] = (_Float16)(go * tanhf(cnew));
            }
        }
        __syncthreads();
    }

    // ---- per-channel linear head on final hidden state ----
    if (tid < BLOC) {
        const _Float16* hT = hbuf[TT & 1];
        const float*    wf = w_fc + (size_t)c * HH;
        float s = b_fc[c];
#pragma unroll 4
        for (int h = 0; h < HH; ++h)
            s += (float)hT[tid * HH + h] * wf[h];
        out[(size_t)(b0 + tid) * CC + c] = s;
    }
}

extern "C" void kernel_launch(void* const* d_in, const int* in_sizes, int n_in,
                              void* d_out, int out_size, void* d_ws, size_t ws_size,
                              hipStream_t stream) {
    const float* x    = (const float*)d_in[0];
    const float* w_ih = (const float*)d_in[1];
    const float* w_hh = (const float*)d_in[2];
    const float* b_ih = (const float*)d_in[3];
    const float* b_hh = (const float*)d_in[4];
    const float* w_fc = (const float*)d_in[5];
    const float* b_fc = (const float*)d_in[6];
    float* out = (float*)d_out;

    lstm_chan_wmma<<<dim3(CC * BSPLIT), dim3(NTHREADS), 0, stream>>>(
        x, w_ih, w_hh, b_ih, b_hh, w_fc, b_fc, out);
}
